// CroCoEncoder_79706003079631
// MI455X (gfx1250) — matrix operand order repaired and checked by reading, not compiled, not run on hardware
//
#include <hip/hip_runtime.h>
#include <hip/hip_bf16.h>
#include <math.h>

// ---------------------------------------------------------------------------
// CroCo ViT encoder forward for MI455X (gfx1250, wave32, WMMA bf16 path)
// B=2, C=3, H=W=512, P=16 -> N=1024 tokens, D=768, NH=12, DH=64, DEPTH=12
// Uses: v_wmma_f32_16x16x32_bf16, global_load_async_to_lds_b128 (ASYNCcnt),
//       split wait counters, workgroup barriers.
// ---------------------------------------------------------------------------

typedef unsigned short u16;
typedef unsigned int   u32;
typedef unsigned long long u64;

typedef __attribute__((ext_vector_type(16))) __bf16 bf16x16;
typedef __attribute__((ext_vector_type(8)))  float  f32x8;
typedef __attribute__((ext_vector_type(8)))  u32    u32x8;

#define TOK_N   1024
#define D_MODEL 768
#define NHEADS  12
#define DHEAD   64
#define ROWS    2048        // B * N

__device__ __forceinline__ u16 f2bf(float f) {
  u32 u = __builtin_bit_cast(u32, f);
  u32 r = (u + 0x7FFFu + ((u >> 16) & 1u)) >> 16;
  return (u16)r;
}

__device__ __forceinline__ void async_ld_b128(u32 lds_off, const void* gptr) {
  asm volatile("global_load_async_to_lds_b128 %0, %1, off"
               :: "v"(lds_off), "v"((u64)(size_t)gptr) : "memory");
}
__device__ __forceinline__ void wait_async0() {
  asm volatile("s_wait_asynccnt 0x0" ::: "memory");
}

// ======================= elementwise f32 -> bf16 convert ====================
__global__ void k_cvt_bf16(const float* __restrict__ in, u16* __restrict__ out, int n) {
  int i = blockIdx.x * blockDim.x + threadIdx.x;
  if (i < n) out[i] = f2bf(in[i]);
}

// ======================= patch gather (im2col for stride-16 conv) ===========
__global__ void k_patch_gather(const float* __restrict__ img, u16* __restrict__ out) {
  int idx = blockIdx.x * blockDim.x + threadIdx.x;   // ROWS*768 threads
  if (idx >= ROWS * D_MODEL) return;
  int k = idx % D_MODEL;
  int r = idx / D_MODEL;
  int b = r >> 10, n = r & 1023;
  int y = n >> 5, x = n & 31;
  int c = k >> 8, rem = k & 255;
  int iy = rem >> 4, ix = rem & 15;
  float v = img[(((size_t)(b * 3 + c) * 512) + (y * 16 + iy)) * 512 + (x * 16 + ix)];
  out[idx] = f2bf(v);
}

// ======================= LayerNorm (wave per row) ===========================
__global__ __launch_bounds__(256) void k_layernorm(
    const float* __restrict__ x, const float* __restrict__ g,
    const float* __restrict__ bta, u16* __restrict__ ob, float* __restrict__ of,
    int mode, float eps) {
  int w = threadIdx.x >> 5, lane = threadIdx.x & 31;
  int row = blockIdx.x * 8 + w;
  const float* xr = x + (size_t)row * D_MODEL;
  float v[24];
  float s = 0.f;
#pragma unroll
  for (int i = 0; i < 24; ++i) { v[i] = xr[lane + 32 * i]; s += v[i]; }
#pragma unroll
  for (int m = 1; m < 32; m <<= 1) s += __shfl_xor(s, m, 32);
  float mean = s * (1.0f / D_MODEL);
  float q = 0.f;
#pragma unroll
  for (int i = 0; i < 24; ++i) { float d = v[i] - mean; q += d * d; }
#pragma unroll
  for (int m = 1; m < 32; m <<= 1) q += __shfl_xor(q, m, 32);
  float rs = rsqrtf(q * (1.0f / D_MODEL) + eps);
#pragma unroll
  for (int i = 0; i < 24; ++i) {
    int c = lane + 32 * i;
    float o = (v[i] - mean) * rs * g[c] + bta[c];
    if (mode == 0) ob[(size_t)row * D_MODEL + c] = f2bf(o);
    else           of[(size_t)row * D_MODEL + c] = o;
  }
}

// ======================= WMMA GEMM: C = A(bf16) * W(bf16)^T + bias ==========
// A: [M,K] row-major bf16 ; W: [N,K] row-major bf16 (i.e. B^T)
// mode 0: Cf = acc+bias ; mode 1: Cf = resid + acc + bias (in-place ok)
// mode 2: Cb = bf16(gelu(acc+bias))
#define BM 128
#define BN 128
#define BK 32
#define LDS_S 48   // padded u16 stride (96B rows -> 16B-aligned b128 chunks)

union AccU { f32x8 v; float f[8]; };

__global__ __launch_bounds__(256, 1) void k_gemm_bf16(
    const u16* __restrict__ A, const u16* __restrict__ W,
    const float* __restrict__ bias, const float* __restrict__ resid,
    float* __restrict__ Cf, u16* __restrict__ Cb,
    int M, int N, int K, int mode) {
  __shared__ __align__(16) u16 As[BM * LDS_S];
  __shared__ __align__(16) u16 Bs[BN * LDS_S];

  int t = threadIdx.x;
  int w = t >> 5, lane = t & 31;
  int m0 = blockIdx.y * BM, n0 = blockIdx.x * BN;

  AccU acc[8];
#pragma unroll
  for (int j = 0; j < 8; ++j)
#pragma unroll
    for (int r = 0; r < 8; ++r) acc[j].f[r] = 0.f;

  // staging geometry: 512 x 16B chunks per tile; thread t moves chunks t, t+256
  int r0 = t >> 2;                 // tile row of first chunk (0..63)
  int c8 = (t & 3) << 3;           // u16 col offset within row (0,8,16,24)
  u32 asb = (u32)(size_t)(&As[0]);
  u32 bsb = (u32)(size_t)(&Bs[0]);
  u32 lds_a0 = asb + (u32)((r0 * LDS_S + c8) << 1);
  u32 lds_a1 = asb + (u32)(((r0 + 64) * LDS_S + c8) << 1);
  u32 lds_b0 = bsb + (u32)((r0 * LDS_S + c8) << 1);
  u32 lds_b1 = bsb + (u32)(((r0 + 64) * LDS_S + c8) << 1);

  int mrow = (w << 4) + (lane & 15);
  int akb = (lane < 16) ? 0 : 16;
  int bkb = (lane < 16) ? 0 : 16;

  for (int k0 = 0; k0 < K; k0 += BK) {
    // async stage A and B tiles straight into LDS (bypasses VGPRs)
    async_ld_b128(lds_a0, A + (size_t)(m0 + r0) * K + k0 + c8);
    async_ld_b128(lds_a1, A + (size_t)(m0 + r0 + 64) * K + k0 + c8);
    async_ld_b128(lds_b0, W + (size_t)(n0 + r0) * K + k0 + c8);
    async_ld_b128(lds_b1, W + (size_t)(n0 + r0 + 64) * K + k0 + c8);
    wait_async0();
    __syncthreads();

    u32x8 au;
#pragma unroll
    for (int v = 0; v < 8; ++v) {
      int kk = akb + (v << 1);
      au[v] = *(const u32*)&As[mrow * LDS_S + kk];
    }
    bf16x16 afrag = __builtin_bit_cast(bf16x16, au);
#pragma unroll
    for (int j = 0; j < 8; ++j) {
      int nrow = (j << 4) + (lane & 15);
      u32x8 bu;
#pragma unroll
      for (int v = 0; v < 8; ++v)
        bu[v] = *(const u32*)&Bs[nrow * LDS_S + bkb + (v << 1)];
      bf16x16 bfrag = __builtin_bit_cast(bf16x16, bu);
      acc[j].v = __builtin_amdgcn_wmma_f32_16x16x32_bf16(
          false, afrag, false, bfrag, (short)0, acc[j].v, false, false);
    }
    __syncthreads();
  }

  int mbase = m0 + (w << 4) + ((lane < 16) ? 0 : 8);
#pragma unroll
  for (int j = 0; j < 8; ++j) {
    int col = n0 + (j << 4) + (lane & 15);
    float bs = bias[col];
#pragma unroll
    for (int r = 0; r < 8; ++r) {
      int row = mbase + r;
      float v = acc[j].f[r] + bs;
      size_t o = (size_t)row * N + col;
      if (mode == 2) {
        float gl = 0.5f * v * (1.0f + erff(v * 0.70710678118f));
        Cb[o] = f2bf(gl);
      } else {
        if (mode == 1) v += resid[o];
        Cf[o] = v;
      }
    }
  }
}

// ======================= RoPE + split/pack q,k,v ============================
__global__ void k_rope_pack(const float* __restrict__ qkv,
                            u16* __restrict__ qb, u16* __restrict__ kb,
                            u16* __restrict__ vb) {
  int idx = blockIdx.x * blockDim.x + threadIdx.x;  // B*NH*N*DH
  if (idx >= 2 * NHEADS * TOK_N * DHEAD) return;
  int d = idx & 63;
  int tmp = idx >> 6;
  int n = tmp & 1023; tmp >>= 10;
  int h = tmp % NHEADS;
  int b = tmp / NHEADS;

  size_t base = ((size_t)(b * TOK_N + n)) * (3 * D_MODEL) + h * DHEAD;
  float qv = qkv[base + d];
  float kv = qkv[base + D_MODEL + d];
  float vv = qkv[base + 2 * D_MODEL + d];

  int y = n >> 5, x = n & 31;
  int half = d >> 5;          // 0: rotate by y ; 1: rotate by x
  int i = d & 31;
  float p = (float)(half ? x : y);
  float theta = p * powf(100.0f, -(float)(i & 15) / 16.0f);
  float c = cosf(theta), s = sinf(theta);
  int pd = (i < 16) ? (d + 16) : (d - 16);
  float sgn = (i < 16) ? -1.0f : 1.0f;
  float qp = qkv[base + pd];
  float kp = qkv[base + D_MODEL + pd];
  float qo = (qv * c + sgn * qp * s) * 0.125f;   // *= DH^-0.5
  float ko = kv * c + sgn * kp * s;

  size_t o = ((size_t)((b * NHEADS + h) * TOK_N + n)) * DHEAD + d;
  qb[o] = f2bf(qo);
  kb[o] = f2bf(ko);
  vb[o] = f2bf(vv);
}

// ======================= flash attention (WMMA bf16) ========================
// grid: (N/64, NH, B), block 128 (4 waves, 16 query rows each)
__global__ __launch_bounds__(128, 1) void k_attn(
    const u16* __restrict__ qg, const u16* __restrict__ kg,
    const u16* __restrict__ vg, u16* __restrict__ og) {
  __shared__ __align__(16) u16 Ks[32 * 64];     // [key][dh]
  __shared__ __align__(16) u16 Vs[64 * 36];     // [dh][key], padded
  __shared__ __align__(16) u16 Ps[4][16 * 32];  // per-wave P tile

  int t = threadIdx.x;
  int w = t >> 5, lane = t & 31;
  int b = blockIdx.z, h = blockIdx.y;
  int q0 = blockIdx.x * 64 + w * 16;

  size_t headoff = ((size_t)(b * NHEADS + h)) * TOK_N * DHEAD;
  const u16* qp = qg + headoff;
  const u16* kp = kg + headoff;
  const u16* vp = vg + headoff;

  // q fragments (16 rows x 64, two K=32 slices)
  int m = lane & 15;
  int akb = (lane < 16) ? 0 : 16;
  u32x8 qa0u, qa1u;
#pragma unroll
  for (int v = 0; v < 8; ++v) {
    int kk = akb + (v << 1);
    qa0u[v] = *(const u32*)(qp + (size_t)(q0 + m) * DHEAD + kk);
    qa1u[v] = *(const u32*)(qp + (size_t)(q0 + m) * DHEAD + 32 + kk);
  }
  bf16x16 qa0 = __builtin_bit_cast(bf16x16, qa0u);
  bf16x16 qa1 = __builtin_bit_cast(bf16x16, qa1u);

  float mrow[8], lrow[8];
  AccU O[4];
#pragma unroll
  for (int r = 0; r < 8; ++r) { mrow[r] = -3.0e38f; lrow[r] = 0.f; }
#pragma unroll
  for (int j = 0; j < 4; ++j)
#pragma unroll
    for (int r = 0; r < 8; ++r) O[j].f[r] = 0.f;

  int keyl = lane & 15;
  int dsel = (lane < 16) ? 0 : 16;
  int kb2  = (lane < 16) ? 0 : 16;

  // K staging geometry: 256 x 16B chunks; thread t moves chunks t, t+128
  int rk = t >> 3;                 // key row 0..15
  int ck = (t & 7) << 3;           // u16 col 0..56
  u32 ksb = (u32)(size_t)(&Ks[0]);
  u32 lds_k0 = ksb + (u32)((rk * 64 + ck) << 1);
  u32 lds_k1 = ksb + (u32)(((rk + 16) * 64 + ck) << 1);

  for (int kt = 0; kt < TOK_N / 32; ++kt) {
    int kb0 = kt * 32;
    // K tile: async straight into LDS
    async_ld_b128(lds_k0, kp + (size_t)(kb0 + rk) * DHEAD + ck);
    async_ld_b128(lds_k1, kp + (size_t)(kb0 + rk + 16) * DHEAD + ck);
    // V tile: b128 load + transpose scatter into LDS
#pragma unroll
    for (int i = 0; i < 2; ++i) {
      int rr = rk + 16 * i;
      uint4 vv = *(const uint4*)(vp + (size_t)(kb0 + rr) * DHEAD + ck);
      u32 e[4] = {vv.x, vv.y, vv.z, vv.w};
#pragma unroll
      for (int p2 = 0; p2 < 4; ++p2) {
        Vs[(ck + 2 * p2 + 0) * 36 + rr] = (u16)(e[p2] & 0xFFFFu);
        Vs[(ck + 2 * p2 + 1) * 36 + rr] = (u16)(e[p2] >> 16);
      }
    }
    wait_async0();
    __syncthreads();

    // S tiles: keys 0..15 (S0), 16..31 (S1); dh reduced in two WMMAs each
    AccU S0, S1;
#pragma unroll
    for (int r = 0; r < 8; ++r) { S0.f[r] = 0.f; S1.f[r] = 0.f; }
    u32x8 b0u, b1u;
#pragma unroll
    for (int v = 0; v < 8; ++v) {
      b0u[v] = *(const u32*)&Ks[keyl * 64 + dsel + (v << 1)];
      b1u[v] = *(const u32*)&Ks[(16 + keyl) * 64 + dsel + (v << 1)];
    }
    S0.v = __builtin_amdgcn_wmma_f32_16x16x32_bf16(false, qa0, false,
             __builtin_bit_cast(bf16x16, b0u), (short)0, S0.v, false, false);
    S1.v = __builtin_amdgcn_wmma_f32_16x16x32_bf16(false, qa0, false,
             __builtin_bit_cast(bf16x16, b1u), (short)0, S1.v, false, false);
#pragma unroll
    for (int v = 0; v < 8; ++v) {
      b0u[v] = *(const u32*)&Ks[keyl * 64 + 32 + dsel + (v << 1)];
      b1u[v] = *(const u32*)&Ks[(16 + keyl) * 64 + 32 + dsel + (v << 1)];
    }
    S0.v = __builtin_amdgcn_wmma_f32_16x16x32_bf16(false, qa1, false,
             __builtin_bit_cast(bf16x16, b0u), (short)0, S0.v, false, false);
    S1.v = __builtin_amdgcn_wmma_f32_16x16x32_bf16(false, qa1, false,
             __builtin_bit_cast(bf16x16, b1u), (short)0, S1.v, false, false);

    // online softmax per row (rows live on 16-lane halves)
#pragma unroll
    for (int r = 0; r < 8; ++r) {
      float rm = fmaxf(S0.f[r], S1.f[r]);
#pragma unroll
      for (int mk = 1; mk < 16; mk <<= 1) rm = fmaxf(rm, __shfl_xor(rm, mk, 32));
      float mnew = fmaxf(mrow[r], rm);
      float alpha = expf(mrow[r] - mnew);
      float p0 = expf(S0.f[r] - mnew);
      float p1 = expf(S1.f[r] - mnew);
      float ps = p0 + p1;
#pragma unroll
      for (int mk = 1; mk < 16; mk <<= 1) ps += __shfl_xor(ps, mk, 32);
      lrow[r] = lrow[r] * alpha + ps;
      mrow[r] = mnew;
#pragma unroll
      for (int j = 0; j < 4; ++j) O[j].f[r] *= alpha;
      int mloc = r + ((lane < 16) ? 0 : 8);
      Ps[w][mloc * 32 + keyl]      = f2bf(p0);
      Ps[w][mloc * 32 + 16 + keyl] = f2bf(p1);
    }
    asm volatile("s_wait_dscnt 0x0" ::: "memory");

    // P fragment (A layout, K = 32 keys)
    u32x8 pau;
#pragma unroll
    for (int v = 0; v < 8; ++v) {
      int kk = akb + (v << 1);
      pau[v] = *(const u32*)&Ps[w][m * 32 + kk];
    }
    bf16x16 pa = __builtin_bit_cast(bf16x16, pau);
#pragma unroll
    for (int j = 0; j < 4; ++j) {
      int dh = (j << 4) + (lane & 15);
      u32x8 bvu;
#pragma unroll
      for (int v = 0; v < 8; ++v)
        bvu[v] = *(const u32*)&Vs[dh * 36 + kb2 + (v << 1)];
      O[j].v = __builtin_amdgcn_wmma_f32_16x16x32_bf16(false, pa, false,
                 __builtin_bit_cast(bf16x16, bvu), (short)0, O[j].v, false, false);
    }
    __syncthreads();
  }

  // finalize: O / l, write bf16 [row, h*64+dh]
#pragma unroll
  for (int j = 0; j < 4; ++j) {
#pragma unroll
    for (int r = 0; r < 8; ++r) {
      int qr = q0 + r + ((lane < 16) ? 0 : 8);
      int col = h * DHEAD + (j << 4) + (lane & 15);
      float ov = O[j].f[r] / lrow[r];
      og[(size_t)(b * TOK_N + qr) * D_MODEL + col] = f2bf(ov);
    }
  }
}

// ======================= pos output (int32 grid) ============================
__global__ void k_pos(int* __restrict__ out) {
  int idx = blockIdx.x * blockDim.x + threadIdx.x;  // B*N*2
  if (idx >= 2 * TOK_N * 2) return;
  int comp = idx & 1;
  int n = (idx >> 1) & 1023;
  int y = n >> 5, x = n & 31;
  out[idx] = comp ? x : y;
}

// ===========================================================================
static inline int cdiv(int a, int b) { return (a + b - 1) / b; }

extern "C" void kernel_launch(void* const* d_in, const int* in_sizes, int n_in,
                              void* d_out, int out_size, void* d_ws, size_t ws_size,
                              hipStream_t stream) {
  const float* image   = (const float*)d_in[0];
  const float* patch_w = (const float*)d_in[2];
  const float* patch_b = (const float*)d_in[3];
  const float* ln1_w   = (const float*)d_in[4];
  const float* ln1_b   = (const float*)d_in[5];
  const float* qkv_w   = (const float*)d_in[6];
  const float* qkv_b   = (const float*)d_in[7];
  const float* proj_w  = (const float*)d_in[8];
  const float* proj_b  = (const float*)d_in[9];
  const float* ln2_w   = (const float*)d_in[10];
  const float* ln2_b   = (const float*)d_in[11];
  const float* fc1_w   = (const float*)d_in[12];
  const float* fc1_b   = (const float*)d_in[13];
  const float* fc2_w   = (const float*)d_in[14];
  const float* fc2_b   = (const float*)d_in[15];
  const float* encn_w  = (const float*)d_in[16];
  const float* encn_b  = (const float*)d_in[17];

  char* ws = (char*)d_ws;
  size_t off = 0;
  auto carve = [&](size_t bytes) -> char* {
    char* p = ws + off;
    off = (off + bytes + 255) & ~(size_t)255;
    return p;
  };

  const size_t nPW  = (size_t)D_MODEL * D_MODEL;            // 589824
  const size_t nQW  = (size_t)12 * 3 * D_MODEL * D_MODEL;   // 21233664
  const size_t nPRW = (size_t)12 * D_MODEL * D_MODEL;       // 7077888
  const size_t nF1W = (size_t)12 * 4 * D_MODEL * D_MODEL;   // 28311552
  const size_t nF2W = nF1W;

  u16* wpatch = (u16*)carve(nPW * 2);
  u16* wqkv   = (u16*)carve(nQW * 2);
  u16* wproj  = (u16*)carve(nPRW * 2);
  u16* wfc1   = (u16*)carve(nF1W * 2);
  u16* wfc2   = (u16*)carve(nF2W * 2);
  u16* Ximg   = (u16*)carve((size_t)ROWS * D_MODEL * 2);
  float* xf   = (float*)carve((size_t)ROWS * D_MODEL * 4);
  float* qkvf = (float*)carve((size_t)ROWS * 3 * D_MODEL * 4);
  u16* h768   = (u16*)carve((size_t)ROWS * D_MODEL * 2);
  u16* h3072  = (u16*)carve((size_t)ROWS * 4 * D_MODEL * 2);
  u16* qbf    = (u16*)carve((size_t)ROWS * D_MODEL * 2);
  u16* kbf    = (u16*)carve((size_t)ROWS * D_MODEL * 2);
  u16* vbf    = (u16*)carve((size_t)ROWS * D_MODEL * 2);
  u16* obf    = (u16*)carve((size_t)ROWS * D_MODEL * 2);

  // --- weight conversion to bf16 ---
  k_cvt_bf16<<<cdiv((int)nPW, 256), 256, 0, stream>>>(patch_w, wpatch, (int)nPW);
  k_cvt_bf16<<<cdiv((int)nQW, 256), 256, 0, stream>>>(qkv_w, wqkv, (int)nQW);
  k_cvt_bf16<<<cdiv((int)nPRW, 256), 256, 0, stream>>>(proj_w, wproj, (int)nPRW);
  k_cvt_bf16<<<cdiv((int)nF1W, 256), 256, 0, stream>>>(fc1_w, wfc1, (int)nF1W);
  k_cvt_bf16<<<cdiv((int)nF2W, 256), 256, 0, stream>>>(fc2_w, wfc2, (int)nF2W);

  // --- patch embed: gather + GEMM ---
  k_patch_gather<<<cdiv(ROWS * D_MODEL, 256), 256, 0, stream>>>(image, Ximg);
  {
    dim3 g(D_MODEL / BN, ROWS / BM);
    k_gemm_bf16<<<g, 256, 0, stream>>>(Ximg, wpatch, patch_b, nullptr, xf, nullptr,
                                       ROWS, D_MODEL, D_MODEL, 0);
  }

  for (int i = 0; i < 12; ++i) {
    // LN1 -> bf16
    k_layernorm<<<ROWS / 8, 256, 0, stream>>>(xf, ln1_w + i * D_MODEL,
        ln1_b + i * D_MODEL, h768, nullptr, 0, 1e-5f);
    // QKV GEMM
    {
      dim3 g((3 * D_MODEL) / BN, ROWS / BM);
      k_gemm_bf16<<<g, 256, 0, stream>>>(h768, wqkv + (size_t)i * 3 * D_MODEL * D_MODEL,
          qkv_b + (size_t)i * 3 * D_MODEL, nullptr, qkvf, nullptr,
          ROWS, 3 * D_MODEL, D_MODEL, 0);
    }
    // RoPE + pack
    k_rope_pack<<<cdiv(2 * NHEADS * TOK_N * DHEAD, 256), 256, 0, stream>>>(
        qkvf, qbf, kbf, vbf);
    // attention
    {
      dim3 g(TOK_N / 64, NHEADS, 2);
      k_attn<<<g, 128, 0, stream>>>(qbf, kbf, vbf, obf);
    }
    // proj + residual
    {
      dim3 g(D_MODEL / BN, ROWS / BM);
      k_gemm_bf16<<<g, 256, 0, stream>>>(obf, wproj + (size_t)i * D_MODEL * D_MODEL,
          proj_b + (size_t)i * D_MODEL, xf, xf, nullptr,
          ROWS, D_MODEL, D_MODEL, 1);
    }
    // LN2 -> bf16
    k_layernorm<<<ROWS / 8, 256, 0, stream>>>(xf, ln2_w + i * D_MODEL,
        ln2_b + i * D_MODEL, h768, nullptr, 0, 1e-5f);
    // fc1 + gelu -> bf16
    {
      dim3 g((4 * D_MODEL) / BN, ROWS / BM);
      k_gemm_bf16<<<g, 256, 0, stream>>>(h768, wfc1 + (size_t)i * 4 * D_MODEL * D_MODEL,
          fc1_b + (size_t)i * 4 * D_MODEL, nullptr, nullptr, h3072,
          ROWS, 4 * D_MODEL, D_MODEL, 2);
    }
    // fc2 + residual
    {
      dim3 g(D_MODEL / BN, ROWS / BM);
      k_gemm_bf16<<<g, 256, 0, stream>>>(h3072, wfc2 + (size_t)i * 4 * D_MODEL * D_MODEL,
          fc2_b + (size_t)i * D_MODEL, xf, xf, nullptr,
          ROWS, D_MODEL, 4 * D_MODEL, 1);
    }
  }

  // final LN (eps 1e-6) -> fp32 straight into d_out
  k_layernorm<<<ROWS / 8, 256, 0, stream>>>(xf, encn_w, encn_b, nullptr,
                                            (float*)d_out, 1, 1e-6f);
  // pos (int32 bits appended after x)
  k_pos<<<cdiv(2 * TOK_N * 2, 256), 256, 0, stream>>>(
      (int*)((float*)d_out + (size_t)ROWS * D_MODEL));
}
